// AlbertEmbeddings_26456998543742
// MI455X (gfx1250) — matrix operand order repaired and checked by reading, not compiled, not run on hardware
//
#include <hip/hip_runtime.h>
#include <hip/hip_bf16.h>
#include <math.h>

// ---------------- problem constants ----------------
#define B_    64
#define L_    128
#define E_    768
#define SEG_  64
#define K_    128          // 2*B nonzero slots per segment
#define F5E   3840         // 5*E concat feature width
#define LNEPS 1e-12f

typedef __attribute__((ext_vector_type(16))) _Float16 v16h;
typedef __attribute__((ext_vector_type(8)))  _Float16 v8h;
typedef __attribute__((ext_vector_type(4)))  _Float16 v4h;
typedef __attribute__((ext_vector_type(8)))  float    v8f;

// k-tile of 64 halves per LDS stage; +8 pad keeps rows 144B (16B-aligned
// fragment chunks) and staggers banks.
#define KT   64
#define KPAD 72

// =====================================================================
// Generic tiled GEMM:  C[M,N] = act( A[M,Kd] * B + bias )
//   - A row-major, arbitrary lda (implicit im2col for convs: lda < Kd)
//   - TRANSB=0: B row-major [Kd,N]; TRANSB=1: B row-major [N,Kd] (A*B^T)
//   - batched via blockIdx.z (sB=0 shares weights)
//   - f32 -> f16 on LDS fill, f32 WMMA accumulation
//   - full interior tiles take a branch-free float4 fill path + prefetch
// =====================================================================
template<int BM, int BN, int WM, int WN, bool TRANSB, bool RELU>
__global__ __launch_bounds__((BM/WM)*(BN/WN)*32)
void gemm_wmma_f16(const float* __restrict__ A,  int lda, long long sA,
                   const float* __restrict__ Bm, int ldb, long long sB,
                   const float* __restrict__ bias,
                   float* __restrict__ C,        int ldc, long long sC,
                   int M, int N, int Kd)
{
    constexpr int WAVES_N  = BN / WN;
    constexpr int NTHREADS = (BM/WM)*(BN/WN)*32;
    constexpr int MT = WM/16, NT = WN/16;

    __shared__ _Float16 As[BM*KPAD];
    __shared__ _Float16 Bs[BN*KPAD];

    const long long z = blockIdx.z;
    const float* Ab = A  + z*sA;
    const float* Bb = Bm + z*sB;
    float*       Cb = C  + z*sC;

    const int tid  = threadIdx.x;
    const int wave = tid >> 5, lane = tid & 31;
    const int wm   = wave / WAVES_N, wn = wave % WAVES_N;
    const int grp  = lane >> 4, l16 = lane & 15;
    const int m0   = blockIdx.x * BM, n0 = blockIdx.y * BN;

    const bool fullM = (m0 + BM <= M);
    const bool fullN = (n0 + BN <= N);

    v8f acc[MT][NT];
#pragma unroll
    for (int i = 0; i < MT; i++)
#pragma unroll
        for (int j = 0; j < NT; j++) { v8f zz = {}; acc[i][j] = zz; }

    for (int k0 = 0; k0 < Kd; k0 += KT) {
        const bool fullK = (k0 + KT <= Kd);
        __syncthreads();                       // protect previous tile reads

        // ---------------- stage A tile: BM x KT ----------------
        if (fullM && fullK) {
            // vectorized: float4 global load -> 4 packed halves (b64 LDS store)
            for (int idx = tid*4; idx < BM*KT; idx += NTHREADS*4) {
                int m = idx >> 6, k = idx & 63;
                const float* src = Ab + (long long)(m0 + m)*lda + k0 + k;
                float4 f = *(const float4*)src;
                v4h h = { (_Float16)f.x, (_Float16)f.y, (_Float16)f.z, (_Float16)f.w };
                *(v4h*)&As[m*KPAD + k] = h;
                if (k0 + KT < Kd)
                    __builtin_prefetch((const void*)(src + KT), 0, 0);
            }
        } else {
            for (int idx = tid; idx < BM*KT; idx += NTHREADS) {
                int m = idx >> 6, k = idx & 63;
                int gm = m0 + m, gk = k0 + k;
                float v = (gm < M && gk < Kd) ? Ab[(long long)gm*lda + gk] : 0.f;
                As[m*KPAD + k] = (_Float16)v;
            }
        }

        // ---------------- stage B tile, stored n-major (Bs[n][k]) ----------------
        if (TRANSB) {
            if (fullN && fullK) {
                for (int idx = tid*4; idx < BN*KT; idx += NTHREADS*4) {
                    int n = idx >> 6, k = idx & 63;
                    const float* src = Bb + (long long)(n0 + n)*ldb + k0 + k;
                    float4 f = *(const float4*)src;
                    v4h h = { (_Float16)f.x, (_Float16)f.y, (_Float16)f.z, (_Float16)f.w };
                    *(v4h*)&Bs[n*KPAD + k] = h;
                    if (k0 + KT < Kd)
                        __builtin_prefetch((const void*)(src + KT), 0, 0);
                }
            } else {
                for (int idx = tid; idx < BN*KT; idx += NTHREADS) {
                    int n = idx >> 6, k = idx & 63;
                    int gn = n0 + n, gk = k0 + k;
                    float v = (gn < N && gk < Kd) ? Bb[(long long)gn*ldb + gk] : 0.f;
                    Bs[n*KPAD + k] = (_Float16)v;
                }
            }
        } else {
            if (fullN && fullK) {
                // coalesced float4 along n; transpose into n-major LDS
                for (int idx = tid*4; idx < KT*BN; idx += NTHREADS*4) {
                    int k = idx / BN, n = idx % BN;
                    const float* src = Bb + (long long)(k0 + k)*ldb + n0 + n;
                    float4 f = *(const float4*)src;
                    Bs[(n+0)*KPAD + k] = (_Float16)f.x;
                    Bs[(n+1)*KPAD + k] = (_Float16)f.y;
                    Bs[(n+2)*KPAD + k] = (_Float16)f.z;
                    Bs[(n+3)*KPAD + k] = (_Float16)f.w;
                    if (k0 + KT < Kd)
                        __builtin_prefetch((const void*)(src + (long long)KT*ldb), 0, 0);
                }
            } else {
                for (int idx = tid; idx < KT*BN; idx += NTHREADS) {
                    int k = idx / BN, n = idx % BN;
                    int gn = n0 + n, gk = k0 + k;
                    float v = (gn < N && gk < Kd) ? Bb[(long long)gk*ldb + gn] : 0.f;
                    Bs[n*KPAD + k] = (_Float16)v;
                }
            }
        }
        __syncthreads();

        // ---------------- 2 x (fragments + WMMA) per stage ----------------
#pragma unroll
        for (int kk = 0; kk < KT; kk += 32) {
            v16h af[MT], bf[NT];
#pragma unroll
            for (int i = 0; i < MT; i++) {
                // A 16x32: lane row m=l16; h=0..7 -> k=kk+grp*8+h, h=8..15 -> k=kk+16+grp*8+(h-8)
                const _Float16* p = &As[(wm*WM + i*16 + l16)*KPAD + kk + grp*8];
                v8h lo = *(const v8h*)p;
                v8h hi = *(const v8h*)(p + 16);
                af[i] = __builtin_shufflevector(lo, hi, 0,1,2,3,4,5,6,7,8,9,10,11,12,13,14,15);
            }
#pragma unroll
            for (int j = 0; j < NT; j++) {
                // B 32x16: lane col n=l16; h=0..7 -> k=kk+grp*8+h, h=8..15 -> k=kk+16+grp*8+(h-8)
                const _Float16* p = &Bs[(wn*WN + j*16 + l16)*KPAD + kk + grp*8];
                v8h lo = *(const v8h*)p;
                v8h hi = *(const v8h*)(p + 16);
                bf[j] = __builtin_shufflevector(lo, hi, 0,1,2,3,4,5,6,7,8,9,10,11,12,13,14,15);
            }
#pragma unroll
            for (int i = 0; i < MT; i++)
#pragma unroll
                for (int j = 0; j < NT; j++)
                    acc[i][j] = __builtin_amdgcn_wmma_f32_16x16x32_f16(
                        false, af[i], false, bf[j], (short)0, acc[i][j], false, false);
        }
    }

    // ---- epilogue: bias + optional ReLU; C layout m = r + 8*grp, n = l16 ----
#pragma unroll
    for (int i = 0; i < MT; i++)
#pragma unroll
        for (int j = 0; j < NT; j++) {
            int gn = n0 + wn*WN + j*16 + l16;
            float bv = (bias != nullptr && gn < N) ? bias[gn] : 0.f;
#pragma unroll
            for (int r = 0; r < 8; r++) {
                int gm = m0 + wm*WM + i*16 + grp*8 + r;
                if (gm < M && gn < N) {
                    float v = acc[i][j][r] + bv;
                    if (RELU) v = fmaxf(v, 0.f);
                    Cb[(long long)gm*ldc + gn] = v;
                }
            }
        }
}

// =====================================================================
// Support kernels (trivial fraction of runtime)
// =====================================================================

// emb = W_word[ids] + W_pos[l] + W_type[tt]
__global__ void base_embed_k(const float* __restrict__ Ww, const float* __restrict__ Wp,
                             const float* __restrict__ Wt, const int* __restrict__ ids,
                             const int* __restrict__ tt, float* __restrict__ emb)
{
    int idx = blockIdx.x*256 + threadIdx.x;
    if (idx >= B_*L_*E_) return;
    int e = idx % E_, l = (idx / E_) % L_, b = idx / (E_*L_);
    int tok = ids[b*L_ + l], ty = tt[b*L_ + l];
    emb[idx] = Ww[(long long)tok*E_ + e] + Wp[l*E_ + e] + Wt[ty*E_ + e];
}

// row-major ordered nonzero compaction of ((tt+am)==segval && lab!=5); pad -1
__global__ void nonzero_k(const int* __restrict__ tt, const int* __restrict__ am,
                          const int* __restrict__ lab, int segval,
                          int* __restrict__ bIdx, int* __restrict__ pIdx)
{
    __shared__ int cnt[B_], off[B_+1];
    int b = threadIdx.x;                       // 64 threads, one per batch row
    int c = 0;
    for (int p = 0; p < L_; p++)
        if (tt[b*L_+p] + am[b*L_+p] == segval && lab[b*L_+p] != 5) c++;
    cnt[b] = c;
    __syncthreads();
    if (b == 0) { int s = 0; for (int i = 0; i < B_; i++) { off[i] = s; s += cnt[i]; } off[B_] = s; }
    __syncthreads();
    int o = off[b];
    for (int p = 0; p < L_; p++)
        if (tt[b*L_+p] + am[b*L_+p] == segval && lab[b*L_+p] != 5) {
            if (o < K_) { bIdx[o] = b; pIdx[o] = p; }
            o++;
        }
    if (b == 0) for (int i = off[B_]; i < K_; i++) { bIdx[i] = -1; pIdx[i] = -1; }
}

// X[i,s,:] = W_word[ids[bIdx[i], lo+s]]
__global__ void gather_seq_k(const float* __restrict__ Ww, const int* __restrict__ ids,
                             const int* __restrict__ bIdx, int lo, float* __restrict__ X)
{
    int idx = blockIdx.x*256 + threadIdx.x;
    if (idx >= K_*SEG_*E_) return;
    int e = idx % E_, s = (idx / E_) % SEG_, i = idx / (E_*SEG_);
    int b = bIdx[i];
    float v = 0.f;
    if (b >= 0) { int tok = ids[b*L_ + lo + s]; v = Ww[(long long)tok*E_ + e]; }
    X[idx] = v;
}

// de[i,t,j,:] = W_word[def_ids[bIdx[i], pIdx[i], t, j]]
__global__ void gather_def_k(const float* __restrict__ Ww, const int* __restrict__ def,
                             const int* __restrict__ bIdx, const int* __restrict__ pIdx,
                             float* __restrict__ de)
{
    int idx = blockIdx.x*256 + threadIdx.x;
    if (idx >= K_*4*20*E_) return;
    int e = idx % E_, j = (idx / E_) % 20, t = (idx / (E_*20)) % 4, i = idx / (E_*20*4);
    int b = bIdx[i];
    float v = 0.f;
    if (b >= 0) {
        int p = pIdx[i];
        int tok = def[((b*L_ + p)*4 + t)*20 + j];
        v = Ww[(long long)tok*E_ + e];
    }
    de[idx] = v;
}

// in-place scaled row softmax; one wave (32 lanes) per row
__global__ void softmax_k(float* __restrict__ buf, int rows, int n, float scale)
{
    int gw = (blockIdx.x*blockDim.x + threadIdx.x) >> 5;
    int lane = threadIdx.x & 31;
    if (gw >= rows) return;
    float* r = buf + (long long)gw * n;
    float m = -3.4e38f;
    for (int j = lane; j < n; j += 32) m = fmaxf(m, r[j]*scale);
    for (int o = 16; o > 0; o >>= 1) m = fmaxf(m, __shfl_xor(m, o, 32));
    float s = 0.f;
    for (int j = lane; j < n; j += 32) { float e = expf(r[j]*scale - m); r[j] = e; s += e; }
    for (int o = 16; o > 0; o >>= 1) s += __shfl_xor(s, o, 32);
    float inv = 1.f / s;
    for (int j = lane; j < n; j += 32) r[j] *= inv;
}

// feat[b, slot*768+oc] = max_t conv[b, t, oc]   (conv already bias+ReLU'd)
__global__ void poolmax_k(const float* __restrict__ conv, int T, int slot,
                          float* __restrict__ feat, int batch)
{
    int idx = blockIdx.x*256 + threadIdx.x;
    if (idx >= batch*E_) return;
    int oc = idx % E_, b = idx / E_;
    const float* p = conv + (long long)b*T*E_ + oc;
    float m = p[0];
    for (int t = 1; t < T; t++) m = fmaxf(m, p[(long long)t*E_]);
    feat[(long long)b*F5E + slot*E_ + oc] = m;
}

// scores over t=0..3, softmax, ctx = mean over the 5 E-chunks of sum_t att*tag_tot
__global__ void ctx_k(const float* __restrict__ pool_s, const float* __restrict__ tag_tot,
                      float* __restrict__ ctxE)
{
    int k = blockIdx.x, tid = threadIdx.x;       // 256 threads
    __shared__ float red[4*256];
    __shared__ float att[4];
    float part[4] = {0.f, 0.f, 0.f, 0.f};
    for (int d = tid; d < F5E; d += 256) {
        float ps = pool_s[(long long)k*F5E + d];
        for (int t = 0; t < 4; t++)
            part[t] += ps * tag_tot[((long long)k*4 + t)*F5E + d];
    }
    for (int t = 0; t < 4; t++) red[t*256 + tid] = part[t];
    __syncthreads();
    for (int s = 128; s > 0; s >>= 1) {
        if (tid < s) for (int t = 0; t < 4; t++) red[t*256 + tid] += red[t*256 + tid + s];
        __syncthreads();
    }
    if (tid == 0) {
        float m = red[0];
        for (int t = 1; t < 4; t++) m = fmaxf(m, red[t*256]);
        float s = 0.f;
        for (int t = 0; t < 4; t++) { float e = expf(red[t*256] - m); att[t] = e; s += e; }
        for (int t = 0; t < 4; t++) att[t] /= s;
    }
    __syncthreads();
    for (int e = tid; e < E_; e += 256) {
        float acc = 0.f;
        for (int t = 0; t < 4; t++) {
            float a = att[t];
            const float* tp = &tag_tot[((long long)k*4 + t)*F5E];
            for (int c = 0; c < 5; c++) acc += a * tp[c*E_ + e];
        }
        ctxE[(long long)k*E_ + e] = acc * 0.2f;
    }
}

// emb[bIdx[i], pIdx[i], :] += ctx[i, :]   ((b,p) unique within a segment)
__global__ void scatter_add_k(const int* __restrict__ bIdx, const int* __restrict__ pIdx,
                              const float* __restrict__ ctx, float* __restrict__ emb)
{
    int idx = blockIdx.x*256 + threadIdx.x;
    if (idx >= K_*E_) return;
    int e = idx % E_, i = idx / E_;
    int b = bIdx[i];
    if (b < 0) return;
    emb[((long long)(b*L_ + pIdx[i]))*E_ + e] += ctx[idx];
}

// per-row LayerNorm into d_out
__global__ void layernorm_k(const float* __restrict__ emb, const float* __restrict__ g,
                            const float* __restrict__ bb, float* __restrict__ out)
{
    int row = blockIdx.x, tid = threadIdx.x;     // 256 threads
    const float* x = emb + (long long)row*E_;
    __shared__ float red[256];
    float s = 0.f;
    for (int e = tid; e < E_; e += 256) s += x[e];
    red[tid] = s; __syncthreads();
    for (int st = 128; st > 0; st >>= 1) { if (tid < st) red[tid] += red[tid+st]; __syncthreads(); }
    float mu = red[0] / (float)E_;
    __syncthreads();
    float v = 0.f;
    for (int e = tid; e < E_; e += 256) { float d = x[e] - mu; v += d*d; }
    red[tid] = v; __syncthreads();
    for (int st = 128; st > 0; st >>= 1) { if (tid < st) red[tid] += red[tid+st]; __syncthreads(); }
    float inv = rsqrtf(red[0] / (float)E_ + LNEPS);
    for (int e = tid; e < E_; e += 256)
        out[(long long)row*E_ + e] = (x[e] - mu) * inv * g[e] + bb[e];
}

// =====================================================================
// Host orchestration
// =====================================================================
#define GEMM(BM,BN,WM,WN,TB,RL, A,lda,sA, Bp,ldb,sB, bias, C,ldc,sC, M,N,Kd, bat)      \
    gemm_wmma_f16<BM,BN,WM,WN,TB,RL>                                                   \
        <<<dim3(((M)+(BM)-1)/(BM), ((N)+(BN)-1)/(BN), (bat)),                          \
           (BM/WM)*(BN/WN)*32, 0, stream>>>(                                           \
            A, lda, (long long)(sA), Bp, ldb, (long long)(sB), bias,                   \
            C, ldc, (long long)(sC), M, N, Kd)

extern "C" void kernel_launch(void* const* d_in, const int* in_sizes, int n_in,
                              void* d_out, int out_size, void* d_ws, size_t ws_size,
                              hipStream_t stream)
{
    (void)in_sizes; (void)n_in; (void)out_size; (void)ws_size;
    // ---- inputs (setup_inputs dict order; params flattened in insertion order) ----
    const int*   ids   = (const int*)d_in[0];
    const int*   tt    = (const int*)d_in[1];
    const int*   am    = (const int*)d_in[2];
    /* d_in[3] input_tag_ids: unused by reference */
    const int*   defid = (const int*)d_in[4];
    const int*   lab   = (const int*)d_in[5];
    const float* Ww    = (const float*)d_in[6];
    const float* Wp    = (const float*)d_in[7];
    const float* Wt    = (const float*)d_in[8];
    /* d_in[9] W_tag: unused by reference */
    const float* lng   = (const float*)d_in[10];
    const float* lnb   = (const float*)d_in[11];
    const float* Wq    = (const float*)d_in[12];
    const float* Wk    = (const float*)d_in[13];
    const float* Wv    = (const float*)d_in[14];
    const float* bq    = (const float*)d_in[15];
    const float* bk    = (const float*)d_in[16];
    const float* bv    = (const float*)d_in[17];
    const float* cswW[5], *cswB[5], *ctwW[5], *ctwB[5];
    for (int f = 0; f < 5; f++) {
        cswW[f] = (const float*)d_in[18 + f];
        cswB[f] = (const float*)d_in[23 + f];
        ctwW[f] = (const float*)d_in[28 + f];
        ctwB[f] = (const float*)d_in[33 + f];
    }
    const float* projTagW = (const float*)d_in[38];   // cnn_proj_w  (3840x3840)
    const float* projTagB = (const float*)d_in[39];
    const float* projSW   = (const float*)d_in[40];   // cnn_s_proj_w
    const float* projSB   = (const float*)d_in[41];
    const float* projW    = (const float*)d_in[42];   // proj_w (768x768)
    const float* projB    = (const float*)d_in[43];
    float* out = (float*)d_out;

    // ---- workspace carve (~174 MB; everything read is written first each call) ----
    float* W = (float*)d_ws;
    size_t o = 0;
    int*   bIdx = (int*)(W + o); o += K_;
    int*   pIdx = (int*)(W + o); o += K_;
    float* EMB  = W + o; o += (size_t)B_*L_*E_;                  // 6.29M
    const size_t ROWS = (size_t)K_*4*20;                         // 10240 (max rows)
    float* BUF0 = W + o; o += ROWS*E_;                           // X / de / attn-out
    float* BUF1 = W + o; o += ROWS*E_;                           // Q / conv scratch
    float* BUF2 = W + o; o += ROWS*E_;                           // K
    float* BUF3 = W + o; o += ROWS*E_;                           // V
    float* SCOR = W + o; o += (size_t)K_*64*64;                  // attn logits
    float* SFEAT= W + o; o += (size_t)K_*F5E;
    float* SPROJ= W + o; o += (size_t)K_*F5E;
    float* TFEAT= W + o; o += (size_t)K_*4*F5E;
    float* TPROJ= W + o; o += (size_t)K_*4*F5E;
    float* CTXE = W + o; o += (size_t)K_*E_;
    float* CTXO = W + o; o += (size_t)K_*E_;

    const float iscale = 1.0f / sqrtf((float)E_);

    // 1) base embeddings
    base_embed_k<<<(B_*L_*E_ + 255)/256, 256, 0, stream>>>(Ww, Wp, Wt, ids, tt, EMB);

    // 2) two context passes
    for (int seg = 0; seg < 2; seg++) {
        const int segval = seg + 1, lo = seg * SEG_;

        nonzero_k<<<1, 64, 0, stream>>>(tt, am, lab, segval, bIdx, pIdx);

        // ---- sentence path: X = word_emb[b_idx, lo:hi] ----
        gather_seq_k<<<(K_*SEG_*E_ + 255)/256, 256, 0, stream>>>(Ww, ids, bIdx, lo, BUF0);
        const int MR = K_*SEG_;                       // 8192 rows
        GEMM(128,128,64,32,false,false, BUF0,E_,0, Wq,E_,0, bq, BUF1,E_,0, MR,E_,E_, 1);
        GEMM(128,128,64,32,false,false, BUF0,E_,0, Wk,E_,0, bk, BUF2,E_,0, MR,E_,E_, 1);
        GEMM(128,128,64,32,false,false, BUF0,E_,0, Wv,E_,0, bv, BUF3,E_,0, MR,E_,E_, 1);
        // scores = Q K^T / sqrt(E), softmax, out = att V   (batch = 128 seqs)
        GEMM(64,64,32,32,true,false, BUF1,E_,SEG_*E_, BUF2,E_,SEG_*E_, (const float*)nullptr,
             SCOR,SEG_,SEG_*SEG_, SEG_,SEG_,E_, K_);
        softmax_k<<<(K_*SEG_ + 7)/8, 256, 0, stream>>>(SCOR, K_*SEG_, SEG_, iscale);
        GEMM(64,128,32,32,false,false, SCOR,SEG_,SEG_*SEG_, BUF3,E_,SEG_*E_, (const float*)nullptr,
             BUF0,E_,SEG_*E_, SEG_,E_,SEG_, K_);     // X_emb -> BUF0
        // sentence CNN: conv-as-GEMM (overlapping-row A, lda=E), fused bias+ReLU, col-max pool
        for (int f = 1; f <= 5; f++) {
            const int T = SEG_ - f + 1;
            GEMM(64,128,32,32,true,true, BUF0,E_,SEG_*E_, cswW[f-1],f*E_,0, cswB[f-1],
                 BUF1,E_,(long long)T*E_, T,E_,f*E_, K_);
            poolmax_k<<<(K_*E_ + 255)/256, 256, 0, stream>>>(BUF1, T, f-1, SFEAT, K_);
        }
        GEMM(128,128,64,32,false,false, SFEAT,F5E,0, projSW,F5E,0, projSB,
             SPROJ,F5E,0, K_,F5E,F5E, 1);            // pool_s

        // ---- tag path: de = W_word[def_ids[b,p]]  (512 seqs x 20 toks) ----
        gather_def_k<<<(K_*4*20*E_ + 255)/256, 256, 0, stream>>>(Ww, defid, bIdx, pIdx, BUF0);
        const int DR = K_*4*20;                       // 10240 rows
        GEMM(128,128,64,32,false,false, BUF0,E_,0, Wq,E_,0, bq, BUF1,E_,0, DR,E_,E_, 1);
        GEMM(128,128,64,32,false,false, BUF0,E_,0, Wk,E_,0, bk, BUF2,E_,0, DR,E_,E_, 1);
        GEMM(128,128,64,32,false,false, BUF0,E_,0, Wv,E_,0, bv, BUF3,E_,0, DR,E_,E_, 1);
        GEMM(32,32,16,32,true,false, BUF1,E_,20*E_, BUF2,E_,20*E_, (const float*)nullptr,
             SCOR,20,400, 20,20,E_, K_*4);
        softmax_k<<<(K_*4*20 + 7)/8, 256, 0, stream>>>(SCOR, K_*4*20, 20, iscale);
        GEMM(32,128,32,32,false,false, SCOR,20,400, BUF3,E_,20*E_, (const float*)nullptr,
             BUF0,E_,20*E_, 20,E_,20, K_*4);         // de attn out -> BUF0
        for (int f = 1; f <= 5; f++) {
            const int T = 20 - f + 1;
            GEMM(32,128,32,32,true,true, BUF0,E_,20*E_, ctwW[f-1],f*E_,0, ctwB[f-1],
                 BUF1,E_,(long long)T*E_, T,E_,f*E_, K_*4);
            poolmax_k<<<(K_*4*E_ + 255)/256, 256, 0, stream>>>(BUF1, T, f-1, TFEAT, K_*4);
        }
        GEMM(128,128,64,32,false,false, TFEAT,F5E,0, projTagW,F5E,0, projTagB,
             TPROJ,F5E,0, K_*4,F5E,F5E, 1);          // tag_tot

        // ---- attention over tags + chunk-mean + final projection + scatter ----
        ctx_k<<<K_, 256, 0, stream>>>(SPROJ, TPROJ, CTXE);
        GEMM(128,128,64,32,false,false, CTXE,E_,0, projW,E_,0, projB,
             CTXO,E_,0, K_,E_,E_, 1);
        scatter_add_k<<<(K_*E_ + 255)/256, 256, 0, stream>>>(bIdx, pIdx, CTXO, EMB);
    }

    // 3) LayerNorm -> d_out
    layernorm_k<<<B_*L_, 256, 0, stream>>>(EMB, lng, lnb, out);
}
